// MIMOSSMKernel_42348377538917
// MI455X (gfx1250) — matrix operand order repaired and checked by reading, not compiled, not run on hardware
//
#include <hip/hip_runtime.h>
#include <hip/hip_bf16.h>

// ---------- types ----------
typedef __attribute__((ext_vector_type(16))) __bf16 v16bf;
typedef __attribute__((ext_vector_type(8)))  __bf16 v8bf;
typedef __attribute__((ext_vector_type(8)))  float  v8f;
typedef unsigned int u32x4 __attribute__((ext_vector_type(4)));
typedef int          i32x8 __attribute__((ext_vector_type(8)));
typedef int          i32x4 __attribute__((ext_vector_type(4)));

#define BATCH   4
#define SEQ     4096
#define DMODEL  768
#define HEADS   8
#define NSTATE  64
#define DHEAD   96
#define DTRANK  48
#define NTOK    (BATCH*SEQ)            // 16384
#define NCHUNK  32
#define CHLEN   (SEQ/NCHUNK)           // 128
#define YTOT    ((size_t)NTOK*DMODEL)  // 12,582,912

// ---------- helpers ----------
__device__ __forceinline__ unsigned short f2bf(float f) {
    unsigned int u = __float_as_uint(f);
    u = (u + 0x7FFFu + ((u >> 16) & 1u)) >> 16;   // round-to-nearest-even
    return (unsigned short)u;
}

__global__ __launch_bounds__(256)
void cvt_f32_bf16(const float* __restrict__ src, unsigned short* __restrict__ dst, int n) {
    int i = blockIdx.x * 256 + threadIdx.x;
    if (i < n) dst[i] = f2bf(src[i]);
}

// ---------- CDNA5 async / TDM staging ----------
#define LDSROW 40   // padded bf16 elems per LDS row (80B): 64B data + 16B pad

// Per-lane async global->LDS copy of a 128x32 bf16 tile (two 16B chunks/thread).
__device__ __forceinline__ void stage_tile_async(const unsigned short* __restrict__ base,
                                                 unsigned short* dst, int rowBase, int K,
                                                 int k0, int tid)
{
#pragma unroll
    for (int c = tid; c < 512; c += 256) {
        int r  = c >> 2;
        int cc = (c & 3) * 8;
        unsigned lds  = (unsigned)(size_t)(dst + r * LDSROW + cc);           // LDS byte offset (low 32 bits)
        unsigned goff = (unsigned)(((rowBase + r) * K + k0 + cc) * 2);       // byte offset from base
        asm volatile("global_load_async_to_lds_b128 %0, %1, %2"
                     :: "v"(lds), "v"(goff), "s"((unsigned long long)(size_t)base)
                     : "memory");
    }
}

// One TDM descriptor: load 128x32 bf16 tile (row stride K elems) into LDS with
// 16B pad after each 64B row -> LDS row pitch 80B (= LDSROW elems).
__device__ __forceinline__ void tdm_load_tile(const unsigned short* __restrict__ base,
                                              unsigned short* dstLDS, int rowBase, int K, int k0)
{
    unsigned long long gaddr = (unsigned long long)(size_t)(base + (size_t)rowBase * K + k0);
    unsigned lds = (unsigned)(size_t)dstLDS;

    u32x4 g0;
    g0[0] = 1u;                                                // count=1, user descriptor
    g0[1] = lds;                                               // lds_addr (bytes)
    g0[2] = (unsigned)gaddr;                                   // global_addr[31:0]
    g0[3] = (unsigned)((gaddr >> 32) & 0x01FFFFFFu) | (2u << 30); // global_addr[56:32] | type=2

    const unsigned flags = (1u << 16)      // data_size = 1 (2 bytes)
                         | (1u << 20)      // pad_enable
                         | (3u << 22)      // pad_interval: 16 DWORDs (64B)
                         | (3u << 25);     // pad_amount:   4 DWORDs (16B)
    const unsigned td0 = (unsigned)K;      // tensor_dim0 (elems)
    const unsigned td1 = 16384u;           // tensor_dim1 (rows, generous)
    const unsigned tile0 = 32u, tile1 = 128u;
    const unsigned long long st0 = (unsigned long long)K;  // tensor_dim0_stride (elems)

    i32x8 g1;
    g1[0] = (int)flags;
    g1[1] = (int)((td0 & 0xFFFFu) << 16);                         // bits 63:48 = td0 lo16
    g1[2] = (int)(((td0 >> 16) & 0xFFFFu) | ((td1 & 0xFFFFu) << 16));
    g1[3] = (int)(((td1 >> 16) & 0xFFFFu) | (tile0 << 16));       // bits 127:112 = tile_dim0
    g1[4] = (int)(tile1 & 0xFFFFu);                               // tile_dim1, tile_dim2=0
    g1[5] = (int)(unsigned)(st0 & 0xFFFFFFFFu);                   // stride0[31:0]
    g1[6] = (int)(unsigned)((st0 >> 32) & 0xFFFFu);               // stride0[47:32], stride1 lo=0
    g1[7] = 0;

    i32x4 z4 = {0, 0, 0, 0};
    i32x8 z8 = {0, 0, 0, 0, 0, 0, 0, 0};
    __builtin_amdgcn_tensor_load_to_lds(g0, g1, z4, z4, z8, 0);
}

// ---------- WMMA bf16 GEMM:  C[M,N] = A[M,K] * W[N,K]^T  (row-major) ----------
// block tile 128x128, 8 waves (2 in M x 4 in N), K-step 32, double-buffered LDS,
// A staged via per-lane async-to-LDS, W staged via one TDM descriptor per step.
__global__ __launch_bounds__(256)
void gemm_bf16_wmma(const unsigned short* __restrict__ A,
                    const unsigned short* __restrict__ W,
                    float* __restrict__ C, int M, int N, int K)
{
    __shared__ __align__(16) unsigned short sA[2][128 * LDSROW];
    __shared__ __align__(16) unsigned short sW[2][128 * LDSROW];

    const int tid  = threadIdx.x;
    const int wave = tid >> 5;
    const int lane = tid & 31;
    const int l16  = lane & 15;
    const int lhi  = lane >> 4;
    const int wm   = wave >> 2;     // 0..1
    const int wn   = wave & 3;      // 0..3
    const int blockM = blockIdx.y * 128;
    const int blockN = blockIdx.x * 128;

    v8f acc[4][2];
#pragma unroll
    for (int i = 0; i < 4; ++i)
#pragma unroll
        for (int j = 0; j < 2; ++j)
#pragma unroll
            for (int r = 0; r < 8; ++r) acc[i][j][r] = 0.0f;

    // prologue: stage first K tile into buffer 0
    stage_tile_async(A, &sA[0][0], blockM, K, 0, tid);
    if (wave == 0) tdm_load_tile(W, &sW[0][0], blockN, K, 0);
    asm volatile("s_wait_asynccnt 0x0" ::: "memory");
    if (wave == 0) asm volatile("s_wait_tensorcnt 0x0" ::: "memory");
    __syncthreads();

    int buf = 0;
    for (int k0 = 0; k0 < K; k0 += 32) {
        const int nk = k0 + 32;
        if (nk < K) {   // overlap: issue next tile's copies before consuming current
            stage_tile_async(A, &sA[buf ^ 1][0], blockM, K, nk, tid);
            if (wave == 0) tdm_load_tile(W, &sW[buf ^ 1][0], blockN, K, nk);
        }

        // B fragments: b[j] = W[n][k0 + lhi*16 + j], j=0..15
        v16bf bfrag[2];
#pragma unroll
        for (int ns = 0; ns < 2; ++ns) {
            int n = wn * 32 + ns * 16 + l16;
            v8bf lo = *(const v8bf*)(&sW[buf][n * LDSROW + lhi * 16]);
            v8bf hi = *(const v8bf*)(&sW[buf][n * LDSROW + lhi * 16 + 8]);
            bfrag[ns] = __builtin_shufflevector(lo, hi, 0,1,2,3,4,5,6,7,8,9,10,11,12,13,14,15);
        }
        // A fragments: a[j] = A[m][lhi*16 + j], j=0..15
#pragma unroll
        for (int ms = 0; ms < 4; ++ms) {
            int m = wm * 64 + ms * 16 + l16;
            v8bf lo = *(const v8bf*)(&sA[buf][m * LDSROW + lhi * 16]);
            v8bf hi = *(const v8bf*)(&sA[buf][m * LDSROW + lhi * 16 + 8]);
            v16bf afrag = __builtin_shufflevector(lo, hi, 0,1,2,3,4,5,6,7,8,9,10,11,12,13,14,15);
#pragma unroll
            for (int ns = 0; ns < 2; ++ns)
                acc[ms][ns] = __builtin_amdgcn_wmma_f32_16x16x32_bf16(
                    false, afrag, false, bfrag[ns], (short)0, acc[ms][ns], false, false);
        }

        if (nk < K) {
            asm volatile("s_wait_asynccnt 0x0" ::: "memory");
            if (wave == 0) asm volatile("s_wait_tensorcnt 0x0" ::: "memory");
            __syncthreads();
            buf ^= 1;
        }
    }

    // store: c[r] -> row = lhi*8 + r, col = l16
#pragma unroll
    for (int ms = 0; ms < 4; ++ms)
#pragma unroll
        for (int ns = 0; ns < 2; ++ns) {
            int m = blockM + wm * 64 + ms * 16 + lhi * 8;
            int n = blockN + wn * 32 + ns * 16 + l16;
#pragma unroll
            for (int r = 0; r < 8; ++r)
                C[(size_t)(m + r) * N + n] = acc[ms][ns][r];
        }
}

// ---------- prep: u (head means), dt (softplus), x_in -> bf16 ----------
__global__ __launch_bounds__(256)
void prep_kernel(const float* __restrict__ XZ, const float* __restrict__ dt_w,
                 const float* __restrict__ dt_b, const float* __restrict__ dt_bias,
                 float* __restrict__ u_out, float* __restrict__ dt_out,
                 unsigned short* __restrict__ xin_bf)
{
    const int token = blockIdx.x;
    const float* xrow = XZ + (size_t)token * (2 * DMODEL);
    __shared__ float sx[DMODEL];
    __shared__ float sdtw[HEADS * DTRANK];
    const int tid = threadIdx.x;
    for (int i = tid; i < DMODEL; i += 256) {
        float v = xrow[i];
        sx[i] = v;
        xin_bf[(size_t)token * DMODEL + i] = f2bf(v);
    }
    for (int i = tid; i < HEADS * DTRANK; i += 256) sdtw[i] = dt_w[i];
    __syncthreads();

    const int h = tid >> 5, lane = tid & 31;
    // head mean over dh=96
    float s = sx[h * DHEAD + lane] + sx[h * DHEAD + lane + 32] + sx[h * DHEAD + lane + 64];
#pragma unroll
    for (int off = 16; off; off >>= 1) s += __shfl_xor(s, off, 32);
    // dt dot over first 48 features
    float p = sx[lane] * sdtw[h * DTRANK + lane];
    if (lane < 16) p += sx[lane + 32] * sdtw[h * DTRANK + lane + 32];
#pragma unroll
    for (int off = 16; off; off >>= 1) p += __shfl_xor(p, off, 32);
    if (lane == 0) {
        u_out[(size_t)token * HEADS + h] = s * (1.0f / 96.0f);
        float xv = p + dt_b[h] + dt_bias[h];
        dt_out[(size_t)token * HEADS + h] = (xv > 20.0f) ? xv : log1pf(expf(xv));
    }
}

// ---------- complex bilinear A_bar ----------
__device__ __forceinline__ void abar(float dt, float Ar, float Ai, float& abr, float& abi) {
    float dr = dt * Ar, di = dt * Ai;
    float nr = 1.0f + 0.5f * dr, ni = 0.5f * di;
    float er = 1.0f - 0.5f * dr + 1e-8f, ei = -0.5f * di;
    float inv = 1.0f / (er * er + ei * ei);
    abr = (nr * er + ni * ei) * inv;
    abi = (ni * er - nr * ei) * inv;
}

// ---------- scan pass A: per-chunk local state + cumulative A product ----------
__global__ __launch_bounds__(32)
void scan_passA(const float* __restrict__ dtb, const float* __restrict__ ub,
                const float* __restrict__ BC,
                const float* __restrict__ alr, const float* __restrict__ ali,
                float* __restrict__ cstate, float* __restrict__ cprod)
{
    const int blk = blockIdx.x;              // b*256 + h*32 + c
    const int c = blk & 31, h = (blk >> 5) & 7, b = blk >> 8;
    const int lane = threadIdx.x;
    const int n0 = lane, n1 = lane + 32;
    const float Ar0 = -expf(alr[h * NSTATE + n0]), Ai0 = ali[h * NSTATE + n0];
    const float Ar1 = -expf(alr[h * NSTATE + n1]), Ai1 = ali[h * NSTATE + n1];
    float s0r = 0, s0i = 0, s1r = 0, s1i = 0;
    float p0r = 1, p0i = 0, p1r = 1, p1i = 0;
    const size_t tbase = (size_t)b * SEQ + (size_t)c * CHLEN;
    for (int t = 0; t < CHLEN; ++t) {
        size_t tok = tbase + t;
        float dt = dtb[tok * HEADS + h];
        float uu = ub[tok * HEADS + h];
        const float* bcrow = BC + tok * (2 * HEADS * NSTATE) + h * NSTATE;
        float bu0 = bcrow[n0] * uu, bu1 = bcrow[n1] * uu;
        float a0r, a0i, a1r, a1i;
        abar(dt, Ar0, Ai0, a0r, a0i);
        abar(dt, Ar1, Ai1, a1r, a1i);
        float tr;
        tr = a0r * s0r - a0i * s0i + bu0; s0i = a0r * s0i + a0i * s0r; s0r = tr;
        tr = a1r * s1r - a1i * s1i + bu1; s1i = a1r * s1i + a1i * s1r; s1r = tr;
        tr = a0r * p0r - a0i * p0i;       p0i = a0r * p0i + a0i * p0r; p0r = tr;
        tr = a1r * p1r - a1i * p1i;       p1i = a1r * p1i + a1i * p1r; p1r = tr;
    }
    size_t o = ((((size_t)b * HEADS + h) * NCHUNK + c) * NSTATE);
    cstate[(o + n0) * 2] = s0r; cstate[(o + n0) * 2 + 1] = s0i;
    cstate[(o + n1) * 2] = s1r; cstate[(o + n1) * 2 + 1] = s1i;
    cprod[(o + n0) * 2] = p0r;  cprod[(o + n0) * 2 + 1] = p0i;
    cprod[(o + n1) * 2] = p1r;  cprod[(o + n1) * 2 + 1] = p1i;
}

// ---------- scan pass B: sequential carry over chunks; emits final_state ----------
__global__ __launch_bounds__(64)
void scan_passB(const float* __restrict__ cstate, const float* __restrict__ cprod,
                float* __restrict__ carries, float* __restrict__ final_out)
{
    const int blk = blockIdx.x;              // b*8 + h (32 blocks)
    const int h = blk & 7, b = blk >> 3;
    const int n = threadIdx.x;
    float cr = 0, ci = 0;
    for (int c = 0; c < NCHUNK; ++c) {
        size_t o = ((((size_t)b * HEADS + h) * NCHUNK + c) * NSTATE + n) * 2;
        carries[o] = cr; carries[o + 1] = ci;
        float ar = cprod[o], ai = cprod[o + 1];
        float sr = cstate[o], si = cstate[o + 1];
        float nr = ar * cr - ai * ci + sr;
        float ni = ar * ci + ai * cr + si;
        cr = nr; ci = ni;
    }
    size_t fo = (((size_t)b * HEADS + h) * NSTATE + n) * 2;
    final_out[fo] = cr; final_out[fo + 1] = ci;   // complex64 interleaved
}

// ---------- scan pass C: re-scan with carry-in, compute y_t ----------
__global__ __launch_bounds__(32)
void scan_passC(const float* __restrict__ dtb, const float* __restrict__ ub,
                const float* __restrict__ BC,
                const float* __restrict__ alr, const float* __restrict__ ali,
                const float* __restrict__ carries, float* __restrict__ y_out)
{
    const int blk = blockIdx.x;
    const int c = blk & 31, h = (blk >> 5) & 7, b = blk >> 8;
    const int lane = threadIdx.x;
    const int n0 = lane, n1 = lane + 32;
    const float Ar0 = -expf(alr[h * NSTATE + n0]), Ai0 = ali[h * NSTATE + n0];
    const float Ar1 = -expf(alr[h * NSTATE + n1]), Ai1 = ali[h * NSTATE + n1];
    size_t co = ((((size_t)b * HEADS + h) * NCHUNK + c) * NSTATE);
    float s0r = carries[(co + n0) * 2], s0i = carries[(co + n0) * 2 + 1];
    float s1r = carries[(co + n1) * 2], s1i = carries[(co + n1) * 2 + 1];
    const size_t tbase = (size_t)b * SEQ + (size_t)c * CHLEN;
    for (int t = 0; t < CHLEN; ++t) {
        size_t tok = tbase + t;
        float dt = dtb[tok * HEADS + h];
        float uu = ub[tok * HEADS + h];
        const float* bcrow = BC + tok * (2 * HEADS * NSTATE) + h * NSTATE;
        float bu0 = bcrow[n0] * uu, bu1 = bcrow[n1] * uu;
        float a0r, a0i, a1r, a1i;
        abar(dt, Ar0, Ai0, a0r, a0i);
        abar(dt, Ar1, Ai1, a1r, a1i);
        float tr;
        tr = a0r * s0r - a0i * s0i + bu0; s0i = a0r * s0i + a0i * s0r; s0r = tr;
        tr = a1r * s1r - a1i * s1i + bu1; s1i = a1r * s1i + a1i * s1r; s1r = tr;
        // y_t = sum_n C[n] * Re(state[n])   (C_sel is real)
        float val = bcrow[HEADS * NSTATE + n0] * s0r + bcrow[HEADS * NSTATE + n1] * s1r;
#pragma unroll
        for (int off = 16; off; off >>= 1) val += __shfl_xor(val, off, 32);
        if (lane == 0) y_out[tok * HEADS + h] = val;
    }
}

// ---------- combine: (y + d*u) broadcast * silu(z) -> bf16 ----------
__global__ __launch_bounds__(256)
void combine_kernel(const float* __restrict__ y_scan, const float* __restrict__ ub,
                    const float* __restrict__ d_param, const float* __restrict__ XZ,
                    unsigned short* __restrict__ Ybf)
{
    size_t idx = (size_t)blockIdx.x * 256 + threadIdx.x;
    if (idx >= YTOT) return;
    size_t tok = idx / DMODEL;
    int d = (int)(idx % DMODEL);
    int h = d / DHEAD;
    float yv = y_scan[tok * HEADS + h] + d_param[h] * ub[tok * HEADS + h];
    float z = XZ[tok * (2 * DMODEL) + DMODEL + d];
    float sig = 1.0f / (1.0f + expf(-z));
    Ybf[idx] = f2bf(yv * z * sig);
}

// ---------- host ----------
extern "C" void kernel_launch(void* const* d_in, const int* in_sizes, int n_in,
                              void* d_out, int out_size, void* d_ws, size_t ws_size,
                              hipStream_t stream) {
    const float* x        = (const float*)d_in[0];
    const float* w_in     = (const float*)d_in[1];
    const float* dt_w     = (const float*)d_in[2];
    const float* dt_b     = (const float*)d_in[3];
    const float* b_w      = (const float*)d_in[4];
    const float* c_w      = (const float*)d_in[5];
    const float* alr      = (const float*)d_in[6];
    const float* ali      = (const float*)d_in[7];
    const float* d_param  = (const float*)d_in[8];
    const float* dt_bias  = (const float*)d_in[9];
    const float* w_out    = (const float*)d_in[10];
    (void)in_sizes; (void)n_in; (void)out_size; (void)ws_size;

    char* ws = (char*)d_ws;
    size_t off = 0;
    auto take = [&](size_t bytes) { char* p = ws + off; off += (bytes + 255) & ~(size_t)255; return p; };

    unsigned short* Xbf   = (unsigned short*)take((size_t)NTOK * DMODEL * 2);
    unsigned short* Winbf = (unsigned short*)take((size_t)2 * DMODEL * DMODEL * 2);
    unsigned short* BCwbf = (unsigned short*)take((size_t)1024 * DMODEL * 2);
    unsigned short* Wobf  = (unsigned short*)take((size_t)DMODEL * DMODEL * 2);
    float* XZ    = (float*)take((size_t)NTOK * 2 * DMODEL * 4);
    float* BC    = (float*)take((size_t)NTOK * 1024 * 4);
    unsigned short* Xinbf = (unsigned short*)take((size_t)NTOK * DMODEL * 2);
    float* U     = (float*)take((size_t)NTOK * HEADS * 4);
    float* DT    = (float*)take((size_t)NTOK * HEADS * 4);
    float* Yscan = (float*)take((size_t)NTOK * HEADS * 4);
    unsigned short* Ybf = (unsigned short*)take((size_t)NTOK * DMODEL * 2);
    float* CS    = (float*)take((size_t)BATCH * HEADS * NCHUNK * NSTATE * 2 * 4);
    float* CP    = (float*)take((size_t)BATCH * HEADS * NCHUNK * NSTATE * 2 * 4);
    float* CAR   = (float*)take((size_t)BATCH * HEADS * NCHUNK * NSTATE * 2 * 4);

    float* y_out = (float*)d_out;
    float* fstate_out = y_out + YTOT;

    // 1) convert weights + activations to bf16
    cvt_f32_bf16<<<(NTOK * DMODEL + 255) / 256, 256, 0, stream>>>(x, Xbf, NTOK * DMODEL);
    cvt_f32_bf16<<<(2 * DMODEL * DMODEL + 255) / 256, 256, 0, stream>>>(w_in, Winbf, 2 * DMODEL * DMODEL);
    cvt_f32_bf16<<<(512 * DMODEL + 255) / 256, 256, 0, stream>>>(b_w, BCwbf, 512 * DMODEL);
    cvt_f32_bf16<<<(512 * DMODEL + 255) / 256, 256, 0, stream>>>(c_w, BCwbf + (size_t)512 * DMODEL, 512 * DMODEL);
    cvt_f32_bf16<<<(DMODEL * DMODEL + 255) / 256, 256, 0, stream>>>(w_out, Wobf, DMODEL * DMODEL);

    // 2) XZ = X @ W_in^T   (16384 x 1536, K=768)
    gemm_bf16_wmma<<<dim3(1536 / 128, NTOK / 128), 256, 0, stream>>>(Xbf, Winbf, XZ, NTOK, 1536, DMODEL);

    // 3) u, dt, x_in -> bf16
    prep_kernel<<<NTOK, 256, 0, stream>>>(XZ, dt_w, dt_b, dt_bias, U, DT, Xinbf);

    // 4) BC = x_in @ [b_w; c_w]^T   (16384 x 1024, K=768)
    gemm_bf16_wmma<<<dim3(1024 / 128, NTOK / 128), 256, 0, stream>>>(Xinbf, BCwbf, BC, NTOK, 1024, DMODEL);

    // 5-7) chunked parallel scan
    scan_passA<<<BATCH * HEADS * NCHUNK, 32, 0, stream>>>(DT, U, BC, alr, ali, CS, CP);
    scan_passB<<<BATCH * HEADS, 64, 0, stream>>>(CS, CP, CAR, fstate_out);
    scan_passC<<<BATCH * HEADS * NCHUNK, 32, 0, stream>>>(DT, U, BC, alr, ali, CAR, Yscan);

    // 8) gated combine -> bf16
    combine_kernel<<<(int)((YTOT + 255) / 256), 256, 0, stream>>>(Yscan, U, d_param, XZ, Ybf);

    // 9) out = Y @ W_out^T  (16384 x 768, K=768) -> f32 directly into d_out
    gemm_bf16_wmma<<<dim3(DMODEL / 128, NTOK / 128), 256, 0, stream>>>(Ybf, Wobf, y_out, NTOK, DMODEL, DMODEL);
}
